// _RMSNormFP8FusedAddModel_20968030339169
// MI455X (gfx1250) — 8-bit matrix operand order repaired and checked by reading, compiled, not run on hardware
//
#include <hip/hip_runtime.h>

#define N_TOKENS 8192
#define D_IN     4096
#define D_OUT    12288
#define EPSF     1e-5f
#define FP8_MAXF 448.0f

#define TILE_M 128
#define TILE_N 128
#define TILE_K 128
#define LDSPITCH 144

typedef __attribute__((ext_vector_type(16))) int   v16i;
typedef __attribute__((ext_vector_type(8)))  float v8f;
typedef __attribute__((ext_vector_type(4)))  int   v4i;

#if __has_builtin(__builtin_amdgcn_global_load_async_to_lds_b128)
#define HAVE_ASYNC_LDS 1
typedef __attribute__((address_space(1))) v4i GLOBAL_V4I;
typedef __attribute__((address_space(3))) v4i LDS_V4I;
__device__ __forceinline__ void async_copy16(const void* g, void* l) {
    __builtin_amdgcn_global_load_async_to_lds_b128(
        (GLOBAL_V4I*)g, (LDS_V4I*)l, 0, 0);
}
#else
#define HAVE_ASYNC_LDS 0
#endif

#if __has_builtin(__builtin_amdgcn_s_wait_asynccnt)
#define WAIT_ASYNC(n) __builtin_amdgcn_s_wait_asynccnt(n)
#else
#define WAIT_ASYNC(n) asm volatile("s_wait_asynccnt " #n)
#endif

__device__ inline unsigned char f32_to_e4m3_sw(float f) {
    float a = fabsf(f);
    a = fminf(a, FP8_MAXF);
    unsigned int sign = (__float_as_uint(f) >> 24) & 0x80u;
    if (!(a >= 0.0f)) return (unsigned char)(sign | 0x7Fu);
    unsigned char r;
    if (a < 0.015625f) {
        int m = __float2int_rn(a * 512.0f);
        r = (unsigned char)((m >= 8) ? 0x08 : m);
    } else {
        unsigned int x = __float_as_uint(a);
        unsigned int lsb = (x >> 20) & 1u;
        x += 0x7FFFFu + lsb;
        int e = (int)(x >> 23) - 127;
        unsigned int m = (x >> 20) & 7u;
        if (e > 8) { e = 8; m = 6u; }
        r = (unsigned char)(((e + 7) << 3) | m);
    }
    return (unsigned char)(sign | r);
}

__device__ inline unsigned int pack4_fp8(float a, float b, float c, float d) {
    a = fminf(fmaxf(a, -FP8_MAXF), FP8_MAXF);
    b = fminf(fmaxf(b, -FP8_MAXF), FP8_MAXF);
    c = fminf(fmaxf(c, -FP8_MAXF), FP8_MAXF);
    d = fminf(fmaxf(d, -FP8_MAXF), FP8_MAXF);
#if __has_builtin(__builtin_amdgcn_cvt_pk_fp8_f32)
    int v = 0;
    v = __builtin_amdgcn_cvt_pk_fp8_f32(a, b, v, false);
    v = __builtin_amdgcn_cvt_pk_fp8_f32(c, d, v, true);
    return (unsigned int)v;
#else
    return (unsigned int)f32_to_e4m3_sw(a)
         | ((unsigned int)f32_to_e4m3_sw(b) << 8)
         | ((unsigned int)f32_to_e4m3_sw(c) << 16)
         | ((unsigned int)f32_to_e4m3_sw(d) << 24);
#endif
}

__global__ __launch_bounds__(256)
void quant_weights_kernel(const float* __restrict__ w, unsigned int* __restrict__ q) {
    size_t t = (size_t)blockIdx.x * 256 + threadIdx.x;
    const float4* wv = (const float4*)w;
    float4 a = wv[t];
    q[t] = pack4_fp8(a.x, a.y, a.z, a.w);
}

__global__ __launch_bounds__(256)
void fused_add_rmsnorm_quant_kernel(const float* __restrict__ x,
                                    const float* __restrict__ res,
                                    const float* __restrict__ nw,
                                    const float* __restrict__ input_scale,
                                    float* __restrict__ add_out,
                                    unsigned int* __restrict__ q_out) {
    const int row = blockIdx.x;
    const int t   = threadIdx.x;
    const float4* xr = (const float4*)(x   + (size_t)row * D_IN);
    const float4* rr = (const float4*)(res + (size_t)row * D_IN);

    float4 v[4];
    float ss = 0.0f;
#pragma unroll
    for (int i = 0; i < 4; ++i) {
        int idx = t + 256 * i;
        float4 a = xr[idx];
        float4 b = rr[idx];
        float4 s; s.x = a.x + b.x; s.y = a.y + b.y; s.z = a.z + b.z; s.w = a.w + b.w;
        v[i] = s;
        ss += s.x * s.x + s.y * s.y + s.z * s.z + s.w * s.w;
    }

    __shared__ float red[256];
    red[t] = ss;
    __syncthreads();
#pragma unroll
    for (int s = 128; s > 0; s >>= 1) {
        if (t < s) red[t] += red[t + s];
        __syncthreads();
    }
    const float inv_rms = rsqrtf(red[0] * (1.0f / (float)D_IN) + EPSF);
    const float inv_in  = 1.0f / input_scale[0];

    float4* ao = (float4*)(add_out + (size_t)row * D_IN);
    unsigned int* qo = q_out + (size_t)row * (D_IN / 4);
    const float4* nwv = (const float4*)nw;
#pragma unroll
    for (int i = 0; i < 4; ++i) {
        int idx = t + 256 * i;
        ao[idx] = v[i];
        float4 w = nwv[idx];
        float qa = v[i].x * inv_rms * w.x * inv_in;
        float qb = v[i].y * inv_rms * w.y * inv_in;
        float qc = v[i].z * inv_rms * w.z * inv_in;
        float qd = v[i].w * inv_rms * w.w * inv_in;
        qo[idx] = pack4_fp8(qa, qb, qc, qd);
    }
}


__device__ __forceinline__ v16i load_a_frag(const unsigned char* lds, int rowBase, int lane) {
    const int r  = rowBase + (lane & 15);
    const int hi = (lane >> 4) * 16;
    const unsigned char* p = lds + r * LDSPITCH;
    v16i f;
#pragma unroll
    for (int half = 0; half < 2; ++half) {
        const int kb = half * 64 + hi;
#pragma unroll
        for (int j = 0; j < 4; ++j) {
            unsigned long long q = *(const unsigned long long*)(p + kb + 32 * (j >> 1) + 8 * (j & 1));
            f[half * 8 + 2 * j]     = (int)(unsigned int)q;
            f[half * 8 + 2 * j + 1] = (int)(unsigned int)(q >> 32);
        }
    }
    return f;
}

__device__ __forceinline__ v16i load_b_frag(const unsigned char* lds, int colBase, int lane) {
    const int c  = colBase + (lane & 15);
    const int hi = (lane >> 4) * 16;
    const unsigned char* p = lds + c * LDSPITCH;
    v16i f;
#pragma unroll
    for (int g = 0; g < 4; ++g) {
        int4 v = *(const int4*)(p + g * 32 + hi);
        f[4 * g + 0] = v.x; f[4 * g + 1] = v.y; f[4 * g + 2] = v.z; f[4 * g + 3] = v.w;
    }
    return f;
}

__device__ __forceinline__ void stage_tile(const unsigned char* __restrict__ Aq,
                                           const unsigned char* __restrict__ Bq,
                                           unsigned char* lA, unsigned char* lB,
                                           int bm, int bn, int k0, int t) {
#pragma unroll
    for (int i = 0; i < 4; ++i) {
        int seg = t + 256 * i;
        int r = seg >> 3;
        int c = (seg & 7) * 16;
        const void* ga = Aq + (size_t)(bm + r) * D_IN + k0 + c;
        const void* gb = Bq + (size_t)(bn + r) * D_IN + k0 + c;
        void* la = lA + r * LDSPITCH + c;
        void* lb = lB + r * LDSPITCH + c;
#if HAVE_ASYNC_LDS
        async_copy16(ga, la);
        async_copy16(gb, lb);
#else
        *(int4*)la = *(const int4*)ga;
        *(int4*)lb = *(const int4*)gb;
#endif
    }
}

__global__ __launch_bounds__(256)
void gemm_fp8_kernel(const unsigned char* __restrict__ Aq,
                     const unsigned char* __restrict__ Bq,
                     const float* __restrict__ in_scale,
                     const float* __restrict__ w_scale,
                     float* __restrict__ out) {
    __shared__ __align__(16) unsigned char lA[2][TILE_M * LDSPITCH];
    __shared__ __align__(16) unsigned char lB[2][TILE_N * LDSPITCH];

    const int t    = threadIdx.x;
    const int lane = t & 31;
    const int wave = t >> 5;
    const int wm   = wave & 3;
    const int wn   = wave >> 2;
    const int bm   = blockIdx.y * TILE_M;
    const int bn   = blockIdx.x * TILE_N;

    v8f acc[2][4];
#pragma unroll
    for (int m = 0; m < 2; ++m)
#pragma unroll
        for (int n = 0; n < 4; ++n) {
            v8f z = {0.f, 0.f, 0.f, 0.f, 0.f, 0.f, 0.f, 0.f};
            acc[m][n] = z;
        }

    const int NK = D_IN / TILE_K;

    stage_tile(Aq, Bq, lA[0], lB[0], bm, bn, 0, t);

    for (int k = 0; k < NK; ++k) {
        const int cur = k & 1;
        const int nxt = cur ^ 1;

        if (k + 1 < NK)
            stage_tile(Aq, Bq, lA[nxt], lB[nxt], bm, bn, (k + 1) * TILE_K, t);

#if HAVE_ASYNC_LDS
        if (k + 1 < NK) { WAIT_ASYNC(8); } else { WAIT_ASYNC(0); }
#endif
        __syncthreads();

        v16i afrag[2];
#pragma unroll
        for (int m = 0; m < 2; ++m)
            afrag[m] = load_a_frag(lA[cur], wm * 32 + m * 16, lane);
#pragma unroll
        for (int n = 0; n < 4; ++n) {
            v16i bfrag = load_b_frag(lB[cur], wn * 64 + n * 16, lane);
#pragma unroll
            for (int m = 0; m < 2; ++m) {
                acc[m][n] = __builtin_amdgcn_wmma_f32_16x16x128_fp8_fp8(
                    afrag[m], bfrag, (short)0, acc[m][n],
                     false,  false);
            }
        }
        __syncthreads();
    }

    const float scale = in_scale[0] * w_scale[0];
    const int rbase = bm + wm * 32 + (lane >> 4) * 8;
    const int cbase = bn + wn * 64 + (lane & 15);
#pragma unroll
    for (int m = 0; m < 2; ++m)
#pragma unroll
        for (int n = 0; n < 4; ++n)
#pragma unroll
            for (int r = 0; r < 8; ++r)
                out[(size_t)(rbase + m * 16 + r) * D_OUT + (cbase + n * 16)] =
                    acc[m][n][r] * scale;
}

extern "C" void kernel_launch(void* const* d_in, const int* in_sizes, int n_in,
                              void* d_out, int out_size, void* d_ws, size_t ws_size,
                              hipStream_t stream) {
    const float* x        = (const float*)d_in[0];
    const float* residual = (const float*)d_in[1];
    const float* norm_w   = (const float*)d_in[2];
    const float* w_fp8    = (const float*)d_in[3];
    const float* in_scale = (const float*)d_in[4];
    const float* w_scale  = (const float*)d_in[5];

    float* out     = (float*)d_out;
    float* add_out = (float*)d_out + (size_t)N_TOKENS * D_OUT;

    unsigned char* Aq = (unsigned char*)d_ws;
    unsigned char* Wq = Aq + (size_t)N_TOKENS * D_IN;

    {
        const int total_u32 = (D_OUT * (D_IN / 4));
        quant_weights_kernel<<<total_u32 / 256, 256, 0, stream>>>(
            w_fp8, (unsigned int*)Wq);
    }
    fused_add_rmsnorm_quant_kernel<<<N_TOKENS, 256, 0, stream>>>(
        x, residual, norm_w, in_scale, add_out, (unsigned int*)Aq);

    dim3 grid(D_OUT / TILE_N, N_TOKENS / TILE_M);
    gemm_fp8_kernel<<<grid, 256, 0, stream>>>(Aq, Wq, in_scale, w_scale, out);
}
